// LSTM_37477884625366
// MI455X (gfx1250) — matrix operand order repaired and checked by reading, not compiled, not run on hardware
//
#include <hip/hip_runtime.h>
#include <hip/hip_bf16.h>
#include <cstdint>
#include <cstddef>

// ---------------------------------------------------------------------------
// LSTM on gfx1250: bf16 WMMA (v_wmma_f32_16x16x32_bf16), fp32 accumulate.
// Weights (16MB bf16, transposed+concat) stay L2-resident across all 64
// sequential steps. Each wave owns a 32(batch) x 16(hidden) tile across all
// four gates: 2 A fragments share 4 B fragments -> 8 WMMAs per 12 b128 loads.
// ---------------------------------------------------------------------------

#define H_DIM  1024
#define G_DIM  4096          // 4 gates * H_DIM
#define BATCHN 256
#define MSTEPS 64
#define XROW   (MSTEPS * H_DIM)   // 65536 floats per batch row of input

typedef __attribute__((ext_vector_type(16))) __bf16 v16bf;
typedef __attribute__((ext_vector_type(8)))  __bf16 v8bf;
typedef __attribute__((ext_vector_type(8)))  float  v8f;

__device__ __forceinline__ unsigned short f2bf(float f) {
  union { float f; unsigned int u; } v; v.f = f;
  unsigned int u = v.u;
  unsigned int r = (u + 0x7FFFu + ((u >> 16) & 1u)) >> 16;  // round-to-nearest-even
  return (unsigned short)r;
}

// A fragment (16-bit 16x32, ISA 7.12.2): lanes 0-15 hold K {0..7,16..23},
// lanes 16-31 hold K {8..15,24..31}; `base` already includes +half*8 elems.
__device__ __forceinline__ v16bf load_afrag(const unsigned short* base, int k) {
  const v8bf lo = *(const v8bf*)(base + k);
  const v8bf hi = *(const v8bf*)(base + k + 16);
  return __builtin_shufflevector(lo, hi,
      0, 1, 2, 3, 4, 5, 6, 7, 8, 9, 10, 11, 12, 13, 14, 15);
}

#define WMMA_BF16(A, B, C) \
  __builtin_amdgcn_wmma_f32_16x16x32_bf16(false, (A), false, (B), (short)0, (C), false, false)

// ---------------------------------------------------------------------------
// Prep kernels
// ---------------------------------------------------------------------------

__global__ void k_f32_to_bf16(const float* __restrict__ in,
                              unsigned short* __restrict__ out, int n4) {
  int i = blockIdx.x * blockDim.x + threadIdx.x;
  if (i < n4) {
    const float4 v = ((const float4*)in)[i];
    ushort4 o;
    o.x = f2bf(v.x); o.y = f2bf(v.y); o.z = f2bf(v.z); o.w = f2bf(v.w);
    ((ushort4*)out)[i] = o;
  }
}

// out[g][f] = W_gate(f, g%1024) in bf16; out is [4096][1024] (transposed concat).
__global__ void k_pack_wT(const float* __restrict__ Wi, const float* __restrict__ Wf,
                          const float* __restrict__ Wo, const float* __restrict__ Wc,
                          unsigned short* __restrict__ out) {
  int idx = blockIdx.x * blockDim.x + threadIdx.x;   // over 4096*1024
  int g = idx >> 10;
  int f = idx & 1023;
  int h = g & 1023;
  const float* W = (g < 1024) ? Wi : (g < 2048) ? Wf : (g < 3072) ? Wo : Wc;
  out[idx] = f2bf(W[(size_t)f * 1024 + h]);
}

__global__ void k_pack_bias(const float* __restrict__ bi, const float* __restrict__ bf,
                            const float* __restrict__ bo, const float* __restrict__ bc,
                            float* __restrict__ out) {
  int i = blockIdx.x * blockDim.x + threadIdx.x;     // 0..4095
  const float* b = (i < 1024) ? bi : (i < 2048) ? bf : (i < 3072) ? bo : bc;
  out[i] = b[i & 1023];
}

__global__ void k_init_state(unsigned short* __restrict__ H0,
                             float* __restrict__ C, int n) {
  int i = blockIdx.x * blockDim.x + threadIdx.x;
  if (i < n) { H0[i] = 0; C[i] = 0.0f; }
}

// ---------------------------------------------------------------------------
// One LSTM timestep. 128 blocks * 4 waves = 512 waves.
// Wave -> (bt: 0..7 batch supertile of 32 rows, ht: 0..63 hidden tile of 16).
// 8 accumulators: acc[t][gate], t = batch half (rows bt*32+t*16 .. +15).
// ---------------------------------------------------------------------------
__global__ __launch_bounds__(128) void k_lstm_step(
    const unsigned short* __restrict__ xb,    // [256][64*1024] bf16
    const unsigned short* __restrict__ WxT,   // [4096][1024]   bf16
    const unsigned short* __restrict__ WhT,   // [4096][1024]   bf16
    const float*          __restrict__ bcat,  // [4096]
    const unsigned short* __restrict__ Hin,   // [256][1024]    bf16
    unsigned short*       __restrict__ Hout,  // [256][1024]    bf16
    float*                __restrict__ Cbuf,  // [256][1024]
    float*                __restrict__ Hf32,  // [256][1024] (d_out)
    int m, int last) {
  const int lane = threadIdx.x & 31;
  const int wave = blockIdx.x * 4 + (threadIdx.x >> 5);
  const int bt   = wave >> 6;     // 0..7  (32-row batch supertile)
  const int ht   = wave & 63;     // 0..63 (16-col hidden tile)
  const int half = lane >> 4;
  const int ln   = lane & 15;

  const int row0 = bt * 32 + ln;        // A frag 0 rows
  const int row1 = bt * 32 + 16 + ln;   // A frag 1 rows

  // A-side per-lane base pointers (chunk offset half*8 baked in)
  const size_t xoff = (size_t)m * H_DIM + half * 8;
  const unsigned short* aX0 = xb  + (size_t)row0 * XROW + xoff;
  const unsigned short* aX1 = xb  + (size_t)row1 * XROW + xoff;
  const unsigned short* aH0 = Hin + (size_t)row0 * H_DIM + half * 8;
  const unsigned short* aH1 = Hin + (size_t)row1 * H_DIM + half * 8;

  // B-side per-lane base pointers (column N = ln of each gate block)
  const int col = ht * 16 + ln;
  const unsigned short* bx[4];
  const unsigned short* bh[4];
#pragma unroll
  for (int g = 0; g < 4; ++g) {
    bx[g] = WxT + ((size_t)(g * 1024 + col)) * H_DIM + half * 8;
    bh[g] = WhT + ((size_t)(g * 1024 + col)) * H_DIM + half * 8;
  }

  v8f acc[2][4];
#pragma unroll
  for (int t = 0; t < 2; ++t)
#pragma unroll
    for (int g = 0; g < 4; ++g) acc[t][g] = (v8f){};

  // ---- x-side GEMM: gates += x[:, m, :] @ Wx -----------------------------
#pragma unroll 2
  for (int k = 0; k < H_DIM; k += 32) {
    const v16bf a0 = load_afrag(aX0, k);
    const v16bf a1 = load_afrag(aX1, k);
#pragma unroll
    for (int g = 0; g < 4; ++g) {
      const v16bf b = load_afrag(bx[g], k);
      acc[0][g] = WMMA_BF16(a0, b, acc[0][g]);
      acc[1][g] = WMMA_BF16(a1, b, acc[1][g]);
    }
  }

  // ---- h-side GEMM: gates += H @ Wh --------------------------------------
#pragma unroll 2
  for (int k = 0; k < H_DIM; k += 32) {
    const v16bf a0 = load_afrag(aH0, k);
    const v16bf a1 = load_afrag(aH1, k);
#pragma unroll
    for (int g = 0; g < 4; ++g) {
      const v16bf b = load_afrag(bh[g], k);
      acc[0][g] = WMMA_BF16(a0, b, acc[0][g]);
      acc[1][g] = WMMA_BF16(a1, b, acc[1][g]);
    }
  }

  // ---- fused gate nonlinearities + cell/hidden update --------------------
  const float Bi = bcat[col];
  const float Bf = bcat[1024 + col];
  const float Bo = bcat[2048 + col];
  const float Bc = bcat[3072 + col];

#pragma unroll
  for (int t = 0; t < 2; ++t) {
#pragma unroll
    for (int r = 0; r < 8; ++r) {
      const int brow = bt * 32 + t * 16 + half * 8 + r;
      const float gi = acc[t][0][r] + Bi;
      const float gf = acc[t][1][r] + Bf;
      const float go = acc[t][2][r] + Bo;
      const float gc = acc[t][3][r] + Bc;
      const float ig = 1.0f / (1.0f + __expf(-gi));
      const float fg = 1.0f / (1.0f + __expf(-gf));
      const float og = 1.0f / (1.0f + __expf(-go));
      const float ct = tanhf(gc);
      const size_t idx = (size_t)brow * H_DIM + col;
      const float c = fg * Cbuf[idx] + ig * ct;
      Cbuf[idx] = c;
      const float h = og * tanhf(c);
      Hout[idx] = f2bf(h);
      if (last) Hf32[idx] = h;
    }
  }
}

// ---------------------------------------------------------------------------
// Host launch
// ---------------------------------------------------------------------------
extern "C" void kernel_launch(void* const* d_in, const int* in_sizes, int n_in,
                              void* d_out, int out_size, void* d_ws, size_t ws_size,
                              hipStream_t stream) {
  (void)in_sizes; (void)n_in; (void)out_size; (void)ws_size;

  const float* x   = (const float*)d_in[0];
  const float* Wxi = (const float*)d_in[1];
  const float* Whi = (const float*)d_in[2];
  const float* bi  = (const float*)d_in[3];
  const float* Wxf = (const float*)d_in[4];
  const float* Whf = (const float*)d_in[5];
  const float* bf  = (const float*)d_in[6];
  const float* Wxo = (const float*)d_in[7];
  const float* Who = (const float*)d_in[8];
  const float* bo  = (const float*)d_in[9];
  const float* Wxc = (const float*)d_in[10];
  const float* Whc = (const float*)d_in[11];
  const float* bc  = (const float*)d_in[12];
  float* out = (float*)d_out;

  // Workspace carve-up (~52 MB total, all offsets 32B-aligned)
  char* ws = (char*)d_ws;
  unsigned short* xb  = (unsigned short*)ws; ws += (size_t)BATCHN * XROW * 2;      // 32 MB
  unsigned short* WxT = (unsigned short*)ws; ws += (size_t)G_DIM * H_DIM * 2;      //  8 MB
  unsigned short* WhT = (unsigned short*)ws; ws += (size_t)G_DIM * H_DIM * 2;      //  8 MB
  unsigned short* Hb0 = (unsigned short*)ws; ws += (size_t)BATCHN * H_DIM * 2;     // .5 MB
  unsigned short* Hb1 = (unsigned short*)ws; ws += (size_t)BATCHN * H_DIM * 2;     // .5 MB
  float* Cbuf = (float*)ws; ws += (size_t)BATCHN * H_DIM * 4;                      //  1 MB
  float* bcat = (float*)ws;                                                        // 16 KB
  unsigned short* Hbuf[2] = { Hb0, Hb1 };

  const int n4 = BATCHN * XROW / 4;
  k_f32_to_bf16<<<n4 / 256, 256, 0, stream>>>(x, xb, n4);
  k_pack_wT<<<(G_DIM * H_DIM) / 256, 256, 0, stream>>>(Wxi, Wxf, Wxo, Wxc, WxT);
  k_pack_wT<<<(G_DIM * H_DIM) / 256, 256, 0, stream>>>(Whi, Whf, Who, Whc, WhT);
  k_pack_bias<<<G_DIM / 256, 256, 0, stream>>>(bi, bf, bo, bc, bcat);
  k_init_state<<<(BATCHN * H_DIM) / 256, 256, 0, stream>>>(Hb0, Cbuf, BATCHN * H_DIM);

  for (int m = 0; m < MSTEPS; ++m) {
    k_lstm_step<<<128, 128, 0, stream>>>(xb, WxT, WhT, bcat,
                                         Hbuf[m & 1], Hbuf[(m + 1) & 1],
                                         Cbuf, out, m, (m == MSTEPS - 1) ? 1 : 0);
  }
}
